// CausalSelfAttention_85117661872378
// MI455X (gfx1250) — compile-verified, matrix operand order checked by reading, not run on hardware
//
#include <hip/hip_runtime.h>
#include <hip/hip_bf16.h>

// ---------------------------------------------------------------------------
// Causal self-attention, MI455X (gfx1250, wave32, WMMA).
// B=2, S=2048, D=768, H=12, HD=64.
// Pipeline: f32->bf16 convert / weight-transpose, fused QKV WMMA GEMM
// (weights staged in LDS via global_load_async_to_lds_b128, double buffered
// with an explicit two-phase loop), flash-attention (online softmax) with
// WMMA for QK^T and P*V, O-proj GEMM.
// All matmuls use v_wmma_f32_16x16x32_bf16 (f32 accumulate).
// ---------------------------------------------------------------------------

typedef __attribute__((ext_vector_type(16))) __bf16   v16bf;
typedef __attribute__((ext_vector_type(8)))  float    v8f;
typedef __attribute__((ext_vector_type(4)))  unsigned u32x4;

constexpr int Bc  = 2;
constexpr int Sq  = 2048;
constexpr int Dm  = 768;
constexpr int Hn  = 12;
constexpr int HDc = 64;
constexpr int Mrows = Bc * Sq;          // 4096 tokens

__device__ __forceinline__ v8f wmma_bf16(v16bf a, v16bf b, v8f c) {
  // D = A(16x32 bf16) * B(32x16 bf16) + C(16x16 f32)
  return __builtin_amdgcn_wmma_f32_16x16x32_bf16(
      /*neg_a=*/false, a, /*neg_b=*/false, b,
      /*c_mod=*/(short)0, c, /*reuse_a=*/false, /*reuse_b=*/false);
}

// Load one 16-element bf16 fragment row for a 32-wide k-window.
// Per ISA 16-bit A layout: lanes 0-15 need k = {half*8..+7, 16+half*8..+7}.
// With p -> start of 32 contiguous bf16, that is two aligned b128 loads.
__device__ __forceinline__ v16bf frag_row32(const __bf16* p, int half) {
  const u32x4* q = (const u32x4*)p;
  union { u32x4 u[2]; v16bf v; } r;
  r.u[0] = q[half];        // k = half*8 .. half*8+7
  r.u[1] = q[2 + half];    // k = 16+half*8 .. +7
  return r.v;
}

// Async global->LDS copy of 16 bytes (ASYNCcnt-tracked).
// vdst VGPR carries the LDS byte address (low 32 bits of the generic address).
__device__ __forceinline__ void async_copy_b128(unsigned lds_off,
                                                const void* gaddr) {
  asm volatile("global_load_async_to_lds_b128 %0, %1, off"
               :: "v"(lds_off), "v"(gaddr) : "memory");
}
__device__ __forceinline__ void wait_asynccnt0() {
  asm volatile("s_wait_asynccnt 0x0" ::: "memory");
}
__device__ __forceinline__ void wait_dscnt0() {
  asm volatile("s_wait_dscnt 0x0" ::: "memory");
}

// ---------------------------------------------------------------------------
// Kernel 0a: f32 -> bf16 elementwise convert (grid-stride).
// ---------------------------------------------------------------------------
__global__ void cvt_f32_to_bf16(const float* __restrict__ src,
                                __bf16* __restrict__ dst, int n) {
  for (int i = blockIdx.x * blockDim.x + threadIdx.x; i < n;
       i += gridDim.x * blockDim.x)
    dst[i] = (__bf16)src[i];
}

// Kernel 0b: W[k][n] (f32, DxD) -> Wt[n][k] (bf16) so B-fragments read
// contiguous k per output column.
__global__ void transpose_cvt(const float* __restrict__ w,
                              __bf16* __restrict__ wt) {
  int idx = blockIdx.x * blockDim.x + threadIdx.x;
  if (idx < Dm * Dm) {
    int n = idx / Dm, k = idx % Dm;
    wt[idx] = (__bf16)w[k * Dm + n];
  }
}

// ---------------------------------------------------------------------------
// GEMM helpers.  Block = 256 threads = 8 waves; wave = 16 rows x 64 cols.
// Block tile 64 rows x 128 cols.  The 128x32 bf16 weight tile for each k-step
// is staged in LDS (row stride padded to 40 elements -> conflict-free
// ds_load_b128) with async-to-LDS copies, double buffered (two-phase loop)
// so the copy for step k+32 overlaps the WMMAs of step k.  A fragments are
// register double-buffered from global.
// ---------------------------------------------------------------------------
#define LDS_BROW 40   // padded row stride (elements) for staged B tile

__device__ __forceinline__ void stage_btile(const __bf16* __restrict__ wt,
                                            int colB, int kk,
                                            __bf16* bufp) {
  // 128 rows x 32 bf16 = 512 x 16B chunks; 256 threads x 2 chunks.
#pragma unroll
  for (int it = 0; it < 2; ++it) {
    int c = threadIdx.x + it * 256;
    int n = c >> 2, cc = c & 3;
    unsigned loff = (unsigned)(uintptr_t)&bufp[n * LDS_BROW + cc * 8];
    async_copy_b128(loff, wt + (size_t)(colB + n) * Dm + kk + cc * 8);
  }
}

// One k-step: issue all 8 ds_load_b128 (4 B fragments) first, then the 4
// WMMAs back-to-back.  The empty asm pins all four fragments live in
// distinct VGPR octets, preventing regalloc from recycling one octet and
// serializing into load-pair -> wait -> wmma chains.
__device__ __forceinline__ void gemm_step(const __bf16* bufp, int nlb, int nl,
                                          int half, v16bf af, v8f acc[4]) {
  v16bf b0 = frag_row32(&bufp[(nlb + 0 * 16 + nl) * LDS_BROW], half);
  v16bf b1 = frag_row32(&bufp[(nlb + 1 * 16 + nl) * LDS_BROW], half);
  v16bf b2 = frag_row32(&bufp[(nlb + 2 * 16 + nl) * LDS_BROW], half);
  v16bf b3 = frag_row32(&bufp[(nlb + 3 * 16 + nl) * LDS_BROW], half);
  asm volatile("" : "+v"(b0), "+v"(b1), "+v"(b2), "+v"(b3));
  acc[0] = wmma_bf16(af, b0, acc[0]);
  acc[1] = wmma_bf16(af, b1, acc[1]);
  acc[2] = wmma_bf16(af, b2, acc[2]);
  acc[3] = wmma_bf16(af, b3, acc[3]);
}

// Full K loop (Dm = 768 = 24 steps of 32 -> 12 two-phase iterations).
__device__ __forceinline__ void gemm_kloop(const __bf16* __restrict__ arow,
                                           const __bf16* __restrict__ wt,
                                           int colB, int nlb, int nl, int half,
                                           __bf16* bs0, __bf16* bs1,
                                           v8f acc[4]) {
  stage_btile(wt, colB, 0, bs0);
  v16bf af = frag_row32(arow, half);

  for (int kk = 0; kk < Dm; kk += 64) {
    // ---- phase 0: compute from bs0 (k=kk), stage k=kk+32 into bs1 ----
    wait_asynccnt0();        // this wave's copies into bs0 are done
    __syncthreads();         // everyone's are; prior reads of bs0 drained
    stage_btile(wt, colB, kk + 32, bs1);
    v16bf afn = frag_row32(arow + kk + 32, half);
    gemm_step(bs0, nlb, nl, half, af, acc);
    af = afn;

    // ---- phase 1: compute from bs1 (k=kk+32), stage k=kk+64 into bs0 ----
    wait_asynccnt0();
    __syncthreads();
    if (kk + 64 < Dm) {
      stage_btile(wt, colB, kk + 64, bs0);
      afn = frag_row32(arow + kk + 64, half);
    }
    gemm_step(bs1, nlb, nl, half, af, acc);
    af = afn;
  }
}

// ---------------------------------------------------------------------------
// Kernel 1: fused QKV projection.  blockIdx.z selects q/k/v.
// Grid: (4096/64, 768/128, 3).
// q: [B,H,S,64] bf16, pre-scaled by 1/sqrt(64)
// k: [B,H,S,64] bf16
// v: stored TRANSPOSED [B,H,64,S] bf16 (so P*V B-fragments load contiguously)
// ---------------------------------------------------------------------------
__global__ __launch_bounds__(256) void qkv_proj(
    const __bf16* __restrict__ xb,
    const __bf16* __restrict__ wqt, const __bf16* __restrict__ wkt,
    const __bf16* __restrict__ wvt,
    const float* __restrict__ bq, const float* __restrict__ bk,
    const float* __restrict__ bv,
    __bf16* __restrict__ qo, __bf16* __restrict__ ko,
    __bf16* __restrict__ vto) {
  __shared__ __attribute__((aligned(16))) __bf16 bs[2][128 * LDS_BROW];

  const int mode = blockIdx.z;
  const __bf16* wt  = (mode == 0) ? wqt : ((mode == 1) ? wkt : wvt);
  const float*  bia = (mode == 0) ? bq  : ((mode == 1) ? bk  : bv);

  const int wave = threadIdx.x >> 5, lane = threadIdx.x & 31;
  const int half = lane >> 4, nl = lane & 15;
  const int row0 = blockIdx.x * 64 + (wave & 3) * 16;
  const int colB = blockIdx.y * 128;            // block's first output col
  const int nlb  = (wave >> 2) * 64;            // wave's col offset in block

  v8f acc[4] = {{}, {}, {}, {}};
  const __bf16* arow = xb + (size_t)(row0 + nl) * Dm;

  gemm_kloop(arow, wt, colB, nlb, nl, half, bs[0], bs[1], acc);

#pragma unroll
  for (int t = 0; t < 4; ++t) {
    const int col = colB + nlb + t * 16 + nl;    // feature index 0..767
    const int h = col >> 6, hd = col & 63;
    const float bb = bia[col];
#pragma unroll
    for (int r = 0; r < 8; ++r) {
      const int row = row0 + r + 8 * half;       // token index 0..4095
      const int b = row >> 11, s = row & 2047;
      const float v = acc[t][r] + bb;
      if (mode == 0)
        qo[(((size_t)(b * Hn + h) * Sq) + s) * HDc + hd] = (__bf16)(v * 0.125f);
      else if (mode == 1)
        ko[(((size_t)(b * Hn + h) * Sq) + s) * HDc + hd] = (__bf16)v;
      else
        vto[(((size_t)(b * Hn + h) * HDc) + hd) * Sq + s] = (__bf16)v;
    }
  }
}

// ---------------------------------------------------------------------------
// Kernel 2: flash attention (causal, online softmax).
// Grid: (B*H, S/128).  Block = 256 threads = 8 waves; wave = 16 query rows.
// Streams 32-key tiles; per tile: 4 WMMA (QK^T) + 4 WMMA (P*V).
// V fragments are loaded before the softmax VALU chain so the global loads
// overlap the exp/shuffle math.  P transposes C-layout -> A-layout through a
// private per-wave LDS tile guarded by s_wait_dscnt (in-order DS per wave,
// no block barrier -> divergent causal trip counts are safe).
// ---------------------------------------------------------------------------
__global__ __launch_bounds__(256) void attn_fwd(
    const __bf16* __restrict__ qg,   // [B,H,S,64]
    const __bf16* __restrict__ kg,   // [B,H,S,64]
    const __bf16* __restrict__ vtg,  // [B,H,64,S]
    __bf16* __restrict__ og) {       // [B,S,768]
  __shared__ __attribute__((aligned(16))) __bf16 plds[8 * 16 * LDS_BROW];

  const int bh = blockIdx.x;
  const int b = bh / Hn, h = bh % Hn;
  const int wave = threadIdx.x >> 5, lane = threadIdx.x & 31;
  const int half = lane >> 4, nl = lane & 15;
  const int q0 = blockIdx.y * 128 + wave * 16;

  const __bf16* qb = qg + (size_t)bh * Sq * HDc;
  const __bf16* kb = kg + (size_t)bh * Sq * HDc;
  const __bf16* vb = vtg + (size_t)bh * HDc * Sq;

  // Q fragments for this wave's 16 rows (A-layout, contiguous k)
  const __bf16* qrow = qb + (size_t)(q0 + nl) * HDc;
  const v16bf qf0 = frag_row32(qrow, half);
  const v16bf qf1 = frag_row32(qrow + 32, half);

  v8f o0 = {}, o1 = {}, o2 = {}, o3 = {};
  float rmax[8], rsum[8];
#pragma unroll
  for (int r = 0; r < 8; ++r) { rmax[r] = -1e30f; rsum[r] = 0.0f; }

  __bf16* pl = plds + wave * (16 * LDS_BROW);  // 16 rows x 32 cols

  for (int kt = 0; kt < q0 + 16; kt += 32) {
    // ---- batch K fragment loads (grouped into one clause) ----
    const __bf16* kr0 = kb + (size_t)(kt + nl) * HDc;
    const __bf16* kr1 = kb + (size_t)(kt + 16 + nl) * HDc;
    v16bf kf0 = frag_row32(kr0, half);
    v16bf kf1 = frag_row32(kr0 + 32, half);
    v16bf kf2 = frag_row32(kr1, half);
    v16bf kf3 = frag_row32(kr1 + 32, half);
    asm volatile("" : "+v"(kf0), "+v"(kf1), "+v"(kf2), "+v"(kf3));

    // ---- scores: S = Q * K^T  (HD split 0..31 / 32..63) ----
    v8f s0 = {}, s1 = {};
    s0 = wmma_bf16(qf0, kf0, s0);
    s0 = wmma_bf16(qf1, kf1, s0);
    s1 = wmma_bf16(qf0, kf2, s1);
    s1 = wmma_bf16(qf1, kf3, s1);

    // ---- V fragments: independent of scores, overlap with softmax math ----
    const v16bf vf0 = frag_row32(vb + (size_t)(0 * 16 + nl) * Sq + kt, half);
    const v16bf vf1 = frag_row32(vb + (size_t)(1 * 16 + nl) * Sq + kt, half);
    const v16bf vf2 = frag_row32(vb + (size_t)(2 * 16 + nl) * Sq + kt, half);
    const v16bf vf3 = frag_row32(vb + (size_t)(3 * 16 + nl) * Sq + kt, half);

    // ---- prefetch next K tile into cache ----
    if (kt + 32 < q0 + 16) {
      __builtin_prefetch(kb + (size_t)(kt + 32 + nl) * HDc, 0, 0);
      __builtin_prefetch(kb + (size_t)(kt + 48 + nl) * HDc, 0, 0);
    }

    // ---- causal mask + online softmax (row lives across lanes in half) ----
    const int j0 = kt + nl, j1 = kt + 16 + nl;
    float p0[8], p1[8];
#pragma unroll
    for (int r = 0; r < 8; ++r) {
      const int i = q0 + r + 8 * half;
      float a0 = (j0 <= i) ? s0[r] : -1e30f;
      float a1 = (j1 <= i) ? s1[r] : -1e30f;
      float m = fmaxf(a0, a1);
      m = fmaxf(m, __shfl_xor(m, 1, 32));
      m = fmaxf(m, __shfl_xor(m, 2, 32));
      m = fmaxf(m, __shfl_xor(m, 4, 32));
      m = fmaxf(m, __shfl_xor(m, 8, 32));
      m = fmaxf(m, rmax[r]);
      const float corr = __expf(rmax[r] - m);
      rmax[r] = m;
      const float e0 = __expf(a0 - m);
      const float e1 = __expf(a1 - m);
      p0[r] = e0; p1[r] = e1;
      float sr = e0 + e1;
      sr += __shfl_xor(sr, 1, 32);
      sr += __shfl_xor(sr, 2, 32);
      sr += __shfl_xor(sr, 4, 32);
      sr += __shfl_xor(sr, 8, 32);
      rsum[r] = rsum[r] * corr + sr;
      o0[r] *= corr; o1[r] *= corr; o2[r] *= corr; o3[r] *= corr;
    }

    // ---- P: C-layout -> A-layout via per-wave LDS tile ----
#pragma unroll
    for (int r = 0; r < 8; ++r) {
      const int m = r + 8 * half;
      pl[m * LDS_BROW + nl]      = (__bf16)p0[r];
      pl[m * LDS_BROW + 16 + nl] = (__bf16)p1[r];
    }
    wait_dscnt0();   // cross-lane LDS RAW within the wave
    const v16bf pf = frag_row32(pl + nl * LDS_BROW, half);

    // ---- O += P * V ----
    o0 = wmma_bf16(pf, vf0, o0);
    o1 = wmma_bf16(pf, vf1, o1);
    o2 = wmma_bf16(pf, vf2, o2);
    o3 = wmma_bf16(pf, vf3, o3);
  }

  // ---- normalize and merge heads into [B,S,768] bf16 ----
#pragma unroll
  for (int r = 0; r < 8; ++r) {
    const float inv = 1.0f / rsum[r];
    const int s = q0 + r + 8 * half;
    __bf16* orow = og + ((size_t)b * Sq + s) * Dm + h * HDc;
    orow[0 * 16 + nl] = (__bf16)(o0[r] * inv);
    orow[1 * 16 + nl] = (__bf16)(o1[r] * inv);
    orow[2 * 16 + nl] = (__bf16)(o2[r] * inv);
    orow[3 * 16 + nl] = (__bf16)(o3[r] * inv);
  }
}

// ---------------------------------------------------------------------------
// Kernel 3: output projection, f32 result to d_out.  Grid: (64, 6).
// ---------------------------------------------------------------------------
__global__ __launch_bounds__(256) void o_proj(
    const __bf16* __restrict__ ab, const __bf16* __restrict__ wot,
    const float* __restrict__ bo, float* __restrict__ out) {
  __shared__ __attribute__((aligned(16))) __bf16 bs[2][128 * LDS_BROW];

  const int wave = threadIdx.x >> 5, lane = threadIdx.x & 31;
  const int half = lane >> 4, nl = lane & 15;
  const int row0 = blockIdx.x * 64 + (wave & 3) * 16;
  const int colB = blockIdx.y * 128;
  const int nlb  = (wave >> 2) * 64;

  v8f acc[4] = {{}, {}, {}, {}};
  const __bf16* arow = ab + (size_t)(row0 + nl) * Dm;

  gemm_kloop(arow, wot, colB, nlb, nl, half, bs[0], bs[1], acc);

#pragma unroll
  for (int t = 0; t < 4; ++t) {
    const int col = colB + nlb + t * 16 + nl;
    const float bb = bo[col];
#pragma unroll
    for (int r = 0; r < 8; ++r) {
      const int row = row0 + r + 8 * half;
      out[(size_t)row * Dm + col] = acc[t][r] + bb;
    }
  }
}

// ---------------------------------------------------------------------------
// Launch
// ---------------------------------------------------------------------------
extern "C" void kernel_launch(void* const* d_in, const int* in_sizes, int n_in,
                              void* d_out, int out_size, void* d_ws,
                              size_t ws_size, hipStream_t stream) {
  const float* x  = (const float*)d_in[0];
  // d_in[1] = mask (unused; causality applied analytically)
  const float* wq = (const float*)d_in[2];
  const float* bq = (const float*)d_in[3];
  const float* wk = (const float*)d_in[4];
  const float* bk = (const float*)d_in[5];
  const float* wv = (const float*)d_in[6];
  const float* bv = (const float*)d_in[7];
  const float* wo = (const float*)d_in[8];
  const float* bo = (const float*)d_in[9];
  float* out = (float*)d_out;

  char* ws = (char*)d_ws;
  size_t off = 0;
  auto alloc = [&](size_t bytes) -> void* {
    void* p = ws + off;
    off = (off + bytes + 255) & ~(size_t)255;
    return p;
  };

  const size_t tokBytes = (size_t)Mrows * Dm * sizeof(__bf16);   // 6.29 MB
  const size_t wBytes   = (size_t)Dm * Dm * sizeof(__bf16);      // 1.18 MB

  __bf16* xb  = (__bf16*)alloc(tokBytes);
  __bf16* wqt = (__bf16*)alloc(wBytes);
  __bf16* wkt = (__bf16*)alloc(wBytes);
  __bf16* wvt = (__bf16*)alloc(wBytes);
  __bf16* wot = (__bf16*)alloc(wBytes);
  __bf16* qb  = (__bf16*)alloc(tokBytes);   // [B,H,S,64]
  __bf16* kb  = (__bf16*)alloc(tokBytes);   // [B,H,S,64]
  __bf16* vtb = (__bf16*)alloc(tokBytes);   // [B,H,64,S]
  __bf16* atb = (__bf16*)alloc(tokBytes);   // [B,S,768]

  // 0) precision conversion / weight transpose
  cvt_f32_to_bf16<<<dim3(4096), dim3(256), 0, stream>>>(x, xb, Mrows * Dm);
  transpose_cvt<<<dim3(2304), dim3(256), 0, stream>>>(wq, wqt);
  transpose_cvt<<<dim3(2304), dim3(256), 0, stream>>>(wk, wkt);
  transpose_cvt<<<dim3(2304), dim3(256), 0, stream>>>(wv, wvt);
  transpose_cvt<<<dim3(2304), dim3(256), 0, stream>>>(wo, wot);

  // 1) fused QKV projection (z = q/k/v)
  qkv_proj<<<dim3(Mrows / 64, Dm / 128, 3), dim3(256), 0, stream>>>(
      xb, wqt, wkt, wvt, bq, bk, bv, qb, kb, vtb);

  // 2) flash attention
  attn_fwd<<<dim3(Bc * Hn, Sq / 128), dim3(256), 0, stream>>>(qb, kb, vtb, atb);

  // 3) output projection
  o_proj<<<dim3(Mrows / 64, Dm / 128), dim3(256), 0, stream>>>(atb, wot, bo, out);
}
